// CausalHolographicQKV_59442347376765
// MI455X (gfx1250) — compile-verified, matrix operand order checked by reading, not run on hardware
//
#include <hip/hip_runtime.h>
#include <hip/hip_bf16.h>
#include <cstdint>
#include <cstddef>

// ---------------------------------------------------------------------------
// CausalHolographicQKV on MI455X (gfx1250)
//
// out = ifft( Ea*Eb*(Rw+Rm+Rb) - Sf*(ma2*mw2 + ma2*mm2 + mb2*mrb2) ).real
// where Ea..Rb are eps-normalized spectra of the 5 linear projections and
// Sf = fft(cumsum(x, axis=1)).  Exactly equivalent to the reference chain
// of bind/unbind/unit_projection ops (all diagonal in Fourier space,
// conjugate-symmetric spectra => .real passes are exact).
// ---------------------------------------------------------------------------

typedef __attribute__((ext_vector_type(16))) _Float16 v16h;
typedef __attribute__((ext_vector_type(8)))  float    v8f;

#define BB    8
#define SSEQ  2048
#define DD    1024
#define MTOT  (BB*SSEQ)        // 16384 rows
#define NWGT  5
#define NTOT  (NWGT*DD)        // 5120 fused output features
#define FN    DD               // FFT length
#define EPSF  1e-8f

// GEMM tiling: 128x256 block tile, 8 waves (2x4), 64x64 per wave (4x4 frags)
#define MB   128
#define NBT  256
#define KB   32
#define LDR  40                // LDS row stride in halfs (32 + 8 pad -> bank spread)

// ---------------------------------------------------------------------------
// fp32 -> f16 conversions
// ---------------------------------------------------------------------------
__global__ void k_cvt_x(const float* __restrict__ x, _Float16* __restrict__ xh, int n) {
  int i = blockIdx.x * blockDim.x + threadIdx.x;
  if (i < n) xh[i] = (_Float16)x[i];
}

struct WPtrs { const float* w[NWGT]; const float* b[NWGT]; };

__global__ void k_cvt_w(WPtrs p, _Float16* __restrict__ wh, float* __restrict__ bias) {
  int i = blockIdx.x * blockDim.x + threadIdx.x;
  const int nw = NWGT * DD * DD;
  if (i < nw) {
    int slot = i >> 20;                 // DD*DD == 2^20
    int rem  = i & (DD * DD - 1);
    wh[i] = (_Float16)p.w[slot][rem];   // row-major [e][d]: B-tile rows are W rows
  } else if (i < nw + NWGT * DD) {
    int j = i - nw;
    bias[j] = p.b[j >> 10][j & (DD - 1)];
  }
}

// ---------------------------------------------------------------------------
// causal prefix sum along the sequence axis (one thread per (b,d) column;
// consecutive threads hit consecutive d -> fully coalesced per step)
// ---------------------------------------------------------------------------
__global__ void k_cumsum(const float* __restrict__ x, float* __restrict__ scum) {
  int t = blockIdx.x * blockDim.x + threadIdx.x;   // 0 .. B*D-1
  int b = t >> 10;
  int d = t & (DD - 1);
  const float* px = x    + (size_t)b * SSEQ * DD + d;
  float*       ps = scum + (size_t)b * SSEQ * DD + d;
  float acc = 0.0f;
  for (int i = 0; i < SSEQ; ++i) {
    acc += px[(size_t)i * DD];
    ps[(size_t)i * DD] = acc;
  }
}

// ---------------------------------------------------------------------------
// Fused GEMM: proj[16384][5120] = xh[16384][1024] @ whT + bias
// 128x256 tile per block, 8 waves (2x4), each wave 4x4 WMMA fragments
// (16 v_wmma per K-step per wave, 1:1 WMMA : ds_load_b128 ratio),
// double-buffered LDS, software-pipelined global loads.
// ---------------------------------------------------------------------------
__global__ __launch_bounds__(256) void k_gemm(const _Float16* __restrict__ xh,
                                              const _Float16* __restrict__ wh,
                                              const float*    __restrict__ bias,
                                              float*          __restrict__ proj) {
  __shared__ __align__(16) _Float16 As[2][MB  * LDR];   // 2 x 10240 B
  __shared__ __align__(16) _Float16 Bs[2][NBT * LDR];   // 2 x 20480 B

  const int tid  = threadIdx.x;
  const int lane = tid & 31;
  const int wave = tid >> 5;
  const int wm   = wave >> 2;            // 0..1 -> 64-row slab
  const int wn   = wave & 3;             // 0..3 -> 64-col slab
  const int rowBase = blockIdx.y * MB;
  const int colBase = blockIdx.x * NBT;

  // A loader: 2 threads per row, 16 halfs (32B) each
  const int lar  = tid >> 1;             // 0..127
  const int las  = (tid & 1) << 4;       // 0 or 16 halfs
  // B loader: 1 thread per row, 32 halfs (64B) each
  const int lbr  = tid;                  // 0..255

  const _Float16* ga = xh + (size_t)(rowBase + lar) * DD + las;
  const _Float16* gb = wh + (size_t)(colBase + lbr) * DD;

  v8f acc[4][4];
#pragma unroll
  for (int i = 0; i < 4; ++i)
#pragma unroll
    for (int j = 0; j < 4; ++j)
      acc[i][j] = {};

  // prologue: stage K-tile 0
  {
    float4 a0 = *(const float4*)(ga);
    float4 a1 = *(const float4*)(ga + 8);
    float4 b0 = *(const float4*)(gb);
    float4 b1 = *(const float4*)(gb + 8);
    float4 b2 = *(const float4*)(gb + 16);
    float4 b3 = *(const float4*)(gb + 24);
    *(float4*)&As[0][lar * LDR + las]     = a0;
    *(float4*)&As[0][lar * LDR + las + 8] = a1;
    *(float4*)&Bs[0][lbr * LDR +  0]      = b0;
    *(float4*)&Bs[0][lbr * LDR +  8]      = b1;
    *(float4*)&Bs[0][lbr * LDR + 16]      = b2;
    *(float4*)&Bs[0][lbr * LDR + 24]      = b3;
  }
  __syncthreads();

  // ISA 7.12.2 16-bit A/B layout: lane&15 = row(col), lane>>4 picks K-half
  const int kh   = (lane >> 4) << 3;     // 0 or 8 halfs
  const int mrow = lane & 15;

  int buf = 0;
  const int NK = DD / KB;                // 32 K-steps
  for (int kk = 0; kk < NK; ++kk) {
    float4 na0, na1, nb0, nb1, nb2, nb3;
    const bool more = (kk + 1) < NK;
    if (more) {
      const _Float16* pa = ga + (kk + 1) * KB;
      const _Float16* pb = gb + (kk + 1) * KB;
      na0 = *(const float4*)(pa);
      na1 = *(const float4*)(pa + 8);
      nb0 = *(const float4*)(pb);
      nb1 = *(const float4*)(pb + 8);
      nb2 = *(const float4*)(pb + 16);
      nb3 = *(const float4*)(pb + 24);
    }

    v16h af[4], bf[4];
#pragma unroll
    for (int i = 0; i < 4; ++i) {
      const _Float16* p = &As[buf][(wm * 64 + i * 16 + mrow) * LDR + kh];
      ((float4*)&af[i])[0] = *(const float4*)(p);        // K = kh..kh+7
      ((float4*)&af[i])[1] = *(const float4*)(p + 16);   // K = kh+16..kh+23
    }
#pragma unroll
    for (int j = 0; j < 4; ++j) {
      const _Float16* p = &Bs[buf][(wn * 64 + j * 16 + mrow) * LDR + kh];
      ((float4*)&bf[j])[0] = *(const float4*)(p);
      ((float4*)&bf[j])[1] = *(const float4*)(p + 16);
    }

#pragma unroll
    for (int i = 0; i < 4; ++i)
#pragma unroll
      for (int j = 0; j < 4; ++j)
        acc[i][j] = __builtin_amdgcn_wmma_f32_16x16x32_f16(
            false, af[i], false, bf[j], (short)0, acc[i][j], false, false);

    __syncthreads();
    if (more) {
      const int nb_ = buf ^ 1;
      *(float4*)&As[nb_][lar * LDR + las]     = na0;
      *(float4*)&As[nb_][lar * LDR + las + 8] = na1;
      *(float4*)&Bs[nb_][lbr * LDR +  0]      = nb0;
      *(float4*)&Bs[nb_][lbr * LDR +  8]      = nb1;
      *(float4*)&Bs[nb_][lbr * LDR + 16]      = nb2;
      *(float4*)&Bs[nb_][lbr * LDR + 24]      = nb3;
    }
    __syncthreads();
    buf ^= 1;
  }

  // epilogue: C/D layout VGPR r, lane L -> m = r + 8*(L>>4), n = L&15
  const int rhalf = (lane >> 4) << 3;
#pragma unroll
  for (int i = 0; i < 4; ++i) {
#pragma unroll
    for (int j = 0; j < 4; ++j) {
      const int col = colBase + wn * 64 + j * 16 + (lane & 15);
      const float bz = bias[col];
      size_t base = (size_t)(rowBase + wm * 64 + i * 16 + rhalf) * NTOT + col;
      v8f c = acc[i][j];
#pragma unroll
      for (int r = 0; r < 8; ++r)
        proj[base + (size_t)r * NTOT] = c[r] + bz;
    }
  }
}

// ---------------------------------------------------------------------------
// Radix-2 Stockham FFT-1024 in LDS (ping-pong), 256 threads, twiddle table.
// Result ends in `a` (10 stages = even number of swaps).
// ---------------------------------------------------------------------------
__device__ __forceinline__ float2 cmulf(float2 a, float2 b) {
  return make_float2(a.x * b.x - a.y * b.y, a.x * b.y + a.y * b.x);
}

template <bool INVERSE>
__device__ void fft1024(float2* __restrict__ a, float2* __restrict__ b,
                        const float2* __restrict__ tw, int tid) {
  float2* src = a;
  float2* dst = b;
  for (int Ns = 1; Ns < FN; Ns <<= 1) {
    __syncthreads();
    const int twm = (FN / 2) / Ns;
#pragma unroll 2
    for (int j = tid; j < FN / 2; j += 256) {
      const int jm = j & (Ns - 1);
      float2 w = tw[jm * twm];
      if (INVERSE) w.y = -w.y;
      const float2 v0 = src[j];
      const float2 v1 = cmulf(w, src[j + FN / 2]);
      const int d0 = ((j - jm) << 1) + jm;           // (j/Ns)*2Ns + j%Ns
      dst[d0]      = make_float2(v0.x + v1.x, v0.y + v1.y);
      dst[d0 + Ns] = make_float2(v0.x - v1.x, v0.y - v1.y);
    }
    float2* t = src; src = dst; dst = t;
  }
  __syncthreads();
}

// unpack two real spectra from one packed complex FFT:
// Fa = (Z(k)+conj(Z(N-k)))/2 ; Fb = (Z(k)-conj(Z(N-k)))/(2i)
__device__ __forceinline__ void unpack2(float2 z, float2 zr, float2& Fa, float2& Fb) {
  Fa = make_float2(0.5f * (z.x + zr.x), 0.5f * (z.y - zr.y));
  Fb = make_float2(0.5f * (z.y + zr.y), 0.5f * (zr.x - z.x));
}

__device__ __forceinline__ void normspec(float2 F, float2& E, float& m2) {
  float mag = sqrtf(F.x * F.x + F.y * F.y);
  float inv = 1.0f / (mag + EPSF);
  E = make_float2(F.x * inv, F.y * inv);
  float s = mag * inv;
  m2 = s * s;
}

// ---------------------------------------------------------------------------
// Per-row spectral kernel: 3 packed forward FFTs + combine + 1 inverse FFT.
// ---------------------------------------------------------------------------
__global__ __launch_bounds__(256) void k_fft(const float* __restrict__ proj,
                                             const float* __restrict__ scum,
                                             float* __restrict__ out) {
  __shared__ float2 bufA[FN];
  __shared__ float2 bufB[FN];
  __shared__ float2 tw[FN / 2];
  __shared__ float2 P[FN];       // Ea*Eb
  __shared__ float2 Racc[FN];    // Rw + Rm
  __shared__ float  ma2[FN];
  __shared__ float  mb2[FN];
  __shared__ float  gacc[FN];

  const int tid = threadIdx.x;
  const size_t row = blockIdx.x;
  const float* pr = proj + row * (size_t)NTOT;    // [a|b|w|m|rb] x 1024
  const float* sr = scum + row * (size_t)FN;

  for (int k = tid; k < FN / 2; k += 256) {
    float s, c;
    __sincosf(-6.283185307179586f * (float)k / (float)FN, &s, &c);
    tw[k] = make_float2(c, s);
  }

  // ---- pass 1: (lin_a) + i (lin_b) ----
  for (int k = tid; k < FN; k += 256)
    bufA[k] = make_float2(pr[k], pr[FN + k]);
  fft1024<false>(bufA, bufB, tw, tid);
  for (int k = tid; k < FN; k += 256) {
    const int kr = (FN - k) & (FN - 1);
    float2 Fa, Fb, Ea, Eb;
    float sa2, sb2;
    unpack2(bufA[k], bufA[kr], Fa, Fb);
    normspec(Fa, Ea, sa2);
    normspec(Fb, Eb, sb2);
    ma2[k] = sa2;
    mb2[k] = sb2;
    P[k] = cmulf(Ea, Eb);
  }
  __syncthreads();

  // ---- pass 2: (lin_w) + i (lin_m) ----
  for (int k = tid; k < FN; k += 256)
    bufA[k] = make_float2(pr[2 * FN + k], pr[3 * FN + k]);
  fft1024<false>(bufA, bufB, tw, tid);
  for (int k = tid; k < FN; k += 256) {
    const int kr = (FN - k) & (FN - 1);
    float2 Fw, Fm, Rw, Rm;
    float sw2, sm2;
    unpack2(bufA[k], bufA[kr], Fw, Fm);
    normspec(Fw, Rw, sw2);
    normspec(Fm, Rm, sm2);
    Racc[k] = make_float2(Rw.x + Rm.x, Rw.y + Rm.y);
    gacc[k] = ma2[k] * (sw2 + sm2);
  }
  __syncthreads();

  // ---- pass 3: (lin_rb) + i (cumsum) ----
  for (int k = tid; k < FN; k += 256)
    bufA[k] = make_float2(pr[4 * FN + k], sr[k]);
  fft1024<false>(bufA, bufB, tw, tid);
  for (int k = tid; k < FN; k += 256) {
    const int kr = (FN - k) & (FN - 1);
    float2 Frb, Fs, Rb;
    float srb2;
    unpack2(bufA[k], bufA[kr], Frb, Fs);
    normspec(Frb, Rb, srb2);
    const float2 R = make_float2(Racc[k].x + Rb.x, Racc[k].y + Rb.y);
    const float  g = gacc[k] + mb2[k] * srb2;
    float2 spec = cmulf(P[k], R);
    spec.x -= g * Fs.x;
    spec.y -= g * Fs.y;
    bufB[k] = spec;                 // bufB is free; bufA still read at kr
  }
  // barrier provided by first stage of the inverse FFT
  fft1024<true>(bufB, bufA, tw, tid);   // result in bufB

  const float inv = 1.0f / (float)FN;
  for (int k = tid; k < FN; k += 256)
    out[row * (size_t)FN + k] = bufB[k].x * inv;
}

// ---------------------------------------------------------------------------
// launch
// ---------------------------------------------------------------------------
extern "C" void kernel_launch(void* const* d_in, const int* in_sizes, int n_in,
                              void* d_out, int out_size, void* d_ws, size_t ws_size,
                              hipStream_t stream) {
  const float* x = (const float*)d_in[0];
  WPtrs wp;
  wp.w[0] = (const float*)d_in[1];  wp.b[0] = (const float*)d_in[2];   // a
  wp.w[1] = (const float*)d_in[3];  wp.b[1] = (const float*)d_in[4];   // b
  wp.w[2] = (const float*)d_in[5];  wp.b[2] = (const float*)d_in[6];   // w
  wp.w[3] = (const float*)d_in[7];  wp.b[3] = (const float*)d_in[8];   // m
  wp.w[4] = (const float*)d_in[9];  wp.b[4] = (const float*)d_in[10];  // rb
  float* out = (float*)d_out;

  uint8_t* ws = (uint8_t*)d_ws;
  size_t off = 0;
  auto carve = [&](size_t bytes) -> void* {
    off = (off + 255) & ~(size_t)255;
    void* p = ws + off;
    off += bytes;
    return p;
  };
  _Float16* xh   = (_Float16*)carve((size_t)MTOT * DD * sizeof(_Float16));
  _Float16* wh   = (_Float16*)carve((size_t)NTOT * DD * sizeof(_Float16));
  float*    bias = (float*)   carve((size_t)NTOT * sizeof(float));
  float*    proj = (float*)   carve((size_t)MTOT * NTOT * sizeof(float));
  float*    scum = (float*)   carve((size_t)MTOT * DD * sizeof(float));

  {
    int n = MTOT * DD;
    k_cvt_x<<<(n + 255) / 256, 256, 0, stream>>>(x, xh, n);
  }
  {
    int n = NWGT * DD * DD + NWGT * DD;
    k_cvt_w<<<(n + 255) / 256, 256, 0, stream>>>(wp, wh, bias);
  }
  k_cumsum<<<(BB * DD) / 256, 256, 0, stream>>>(x, scum);

  dim3 gg(NTOT / NBT, MTOT / MB);     // 20 x 128 tiles
  k_gemm<<<gg, 256, 0, stream>>>(xh, wh, bias, proj);

  k_fft<<<MTOT, 256, 0, stream>>>(proj, scum, out);
}
